// displacement_decoder_amp_5995774345568
// MI455X (gfx1250) — matrix operand order repaired and checked by reading, not compiled, not run on hardware
//
#include <hip/hip_runtime.h>
#include <hip/hip_bf16.h>
#include <math.h>

// ---------------------------------------------------------------------------
// Types for CDNA5 WMMA (wave32, v_wmma_f32_16x16x32_f16)
// ---------------------------------------------------------------------------
typedef _Float16 v16h __attribute__((ext_vector_type(16)));
typedef _Float16 h8   __attribute__((ext_vector_type(8)));
typedef float    v8f  __attribute__((ext_vector_type(8)));

#define PI_F 3.14159265358979323846f

// Problem sizes (match reference)
#define BATCH   8
#define NPTS    16384
#define MROWS   (BATCH * NPTS)      // 131072
#define NNEU    128
#define DIN_PAD 96                  // 68 padded to multiple of 32
#define NLAYERS 6
#define BOX     128
#define SIM     256
#define NCLS    2

// GEMM tiling: block = 256 threads (8 waves); wave owns 64 rows x 16 cols.
#define MTILE_WAVE 64
#define MSUB       4                // 4 x 16-row WMMA subtiles per wave

// ---------------------------------------------------------------------------
// Workspace layout (bytes)
// ---------------------------------------------------------------------------
static const size_t WS_W0P  = 0;                                   // 128*96 f16
static const size_t WS_W1   = WS_W0P + (size_t)128 * DIN_PAD * 2;  // 6*128*128 f16
static const size_t WS_W2   = WS_W1 + (size_t)NLAYERS * 128 * 128 * 2;
static const size_t WS_FEAT = WS_W2 + (size_t)NLAYERS * 128 * 128 * 2;
static const size_t WS_H    = WS_FEAT + (size_t)MROWS * DIN_PAD * 2;
static const size_t WS_T    = WS_H + (size_t)MROWS * 128 * 2;
static const size_t WS_X    = WS_T + (size_t)MROWS * 128 * 2;      // 16*256*256 float2
static const size_t WS_V3   = WS_X + (size_t)16 * SIM * SIM * 8;   // 8*128 f32

// Output layout (float elements, concatenated in reference return order)
static const size_t O_PROJ = 0;                         // complex64 -> 2 floats each
static const size_t O_IM   = O_PROJ + (size_t)2 * BATCH * SIM * SIM;
static const size_t O_PP   = O_IM + (size_t)BATCH * NCLS * SIM * SIM;
static const size_t O_POS  = O_PP + (size_t)BATCH * NPTS * 2;
static const size_t O_RES  = O_POS + (size_t)BATCH * NPTS * 3;
static const size_t O_M    = O_RES + (size_t)BATCH * NPTS * 3;

// ---------------------------------------------------------------------------
// Weight conversion kernels (f32 -> f16, once per launch; tiny)
// ---------------------------------------------------------------------------
__global__ void k_f32_to_f16(const float* __restrict__ src, _Float16* __restrict__ dst, int n) {
    int i = blockIdx.x * blockDim.x + threadIdx.x;
    if (i < n) dst[i] = (_Float16)src[i];
}

__global__ void k_f32_to_f16_pad(const float* __restrict__ src, _Float16* __restrict__ dst,
                                 int rows, int kin, int kout) {
    int i = blockIdx.x * blockDim.x + threadIdx.x;
    if (i >= rows * kout) return;
    int r = i / kout, k = i - r * kout;
    dst[i] = (_Float16)(k < kin ? src[r * kin + k] : 0.0f);
}

// ---------------------------------------------------------------------------
// Positional-encoding + latent feature builder: FEAT[b*N+n][0..95] (f16)
// ---------------------------------------------------------------------------
__global__ void k_feat(const float* __restrict__ cons, const float* __restrict__ z,
                       _Float16* __restrict__ feat) {
    int i = blockIdx.x * blockDim.x + threadIdx.x;
    if (i >= MROWS) return;
    int b = i >> 14;          // /NPTS
    int n = i & (NPTS - 1);
    _Float16* row = feat + (size_t)i * DIN_PAD;
    #pragma unroll
    for (int d = 0; d < 3; ++d) {
        float c = cons[n * 3 + d];
        #pragma unroll
        for (int f = 0; f < 10; ++f) {
            float freq = 64.0f * __powf(1.0f / 64.0f, (float)f / 9.0f);
            float k = c * freq;
            row[d * 20 + f]      = (_Float16)__sinf(k);
            row[d * 20 + 10 + f] = (_Float16)__cosf(k);
        }
    }
    #pragma unroll
    for (int k = 0; k < 8; ++k) row[60 + k] = (_Float16)z[b * 8 + k];
    #pragma unroll
    for (int k = 68; k < DIN_PAD; ++k) row[k] = (_Float16)0.0f;
}

// ---------------------------------------------------------------------------
// WMMA GEMM: Out[M,128] = epilogue(A[M,K] @ W[N=128,K]^T)
//   MODE 0: Out = acc
//   MODE 1: Out = relu(acc)
//   MODE 2: Out = Resid + relu(acc)
// Block = 256 threads (8 waves). Block owns 64 rows; wave w owns cols [16w,16w+16)
// and all 4 x 16-row subtiles, so each B (weight) fragment feeds 4 WMMAs.
// Fragment layouts (16-bit A 16x32, B 32x16): contiguous k = 16*half + e in both; f32 C/D per CDNA5 ISA 7.12.2.
// ---------------------------------------------------------------------------
template <int MODE>
__global__ void k_gemm_wmma(const _Float16* __restrict__ A,
                            const _Float16* __restrict__ W,
                            _Float16* __restrict__ Out,
                            const _Float16* __restrict__ Resid,
                            int K) {
    const int lane = threadIdx.x & 31;
    const int wave = threadIdx.x >> 5;
    const int rowbase = blockIdx.x * MTILE_WAVE;
    const int n0 = wave * 16;
    const int l16 = lane & 15;
    const int half = lane >> 4;

    const _Float16* Arow = A + (size_t)(rowbase + l16) * K;   // A-frag: row = lane&15
    const _Float16* Wrow = W + (size_t)(n0 + l16) * K;        // B-frag: col = lane&15
    const size_t astride = (size_t)16 * K;                    // 16 rows between subtiles

    v8f acc[MSUB];
    #pragma unroll
    for (int mt = 0; mt < MSUB; ++mt) acc[mt] = (v8f){};

    for (int k0 = 0; k0 < K; k0 += 32) {
        // B 32x16 f16: lane col fixed, elem e -> K = k0 + half*16 + e (contiguous 32B)
        v16h bf = *(const v16h*)(Wrow + k0 + half * 16);
        #pragma unroll
        for (int mt = 0; mt < MSUB; ++mt) {
            // A 16x32 f16: elem e -> K = k0+half*16+e (contiguous, matches B)
            const _Float16* p = Arow + (size_t)mt * astride + k0 + half * 16;
            union { v16h v; h8 h[2]; } af;
            af.h[0] = *(const h8*)(p);
            af.h[1] = *(const h8*)(p + 8);
            acc[mt] = __builtin_amdgcn_wmma_f32_16x16x32_f16(
                false, af.v, false, bf, (short)0, acc[mt], false, false);
        }
    }

    // C/D layout: col = lane&15 ; VGPR v -> row = v + 8*half
    #pragma unroll
    for (int mt = 0; mt < MSUB; ++mt) {
        #pragma unroll
        for (int v = 0; v < 8; ++v) {
            int r = rowbase + mt * 16 + v + 8 * half;
            float x = acc[mt][v];
            if (MODE >= 1) x = x > 0.0f ? x : 0.0f;
            if (MODE == 2) x += (float)Resid[(size_t)r * NNEU + n0 + l16];
            Out[(size_t)r * NNEU + n0 + l16] = (_Float16)x;
        }
    }
}

// ---------------------------------------------------------------------------
// Head: res = tanh(h @ Wa^T) @ Wb^T ; pos = cons + res ; proj = (pos @ r_b)[:2]
// ---------------------------------------------------------------------------
__global__ void k_head(const _Float16* __restrict__ H, const float* __restrict__ cons,
                       const float* __restrict__ r, const float* __restrict__ Wa,
                       const float* __restrict__ Wb,
                       float* __restrict__ out_pp, float* __restrict__ out_pos,
                       float* __restrict__ out_res) {
    int i = blockIdx.x * blockDim.x + threadIdx.x;
    if (i >= MROWS) return;
    int b = i >> 14;
    int n = i & (NPTS - 1);
    const _Float16* h = H + (size_t)i * NNEU;
    float a[3];
    #pragma unroll
    for (int j = 0; j < 3; ++j) {
        float s = 0.0f;
        for (int k = 0; k < NNEU; ++k) s += (float)h[k] * Wa[j * NNEU + k];
        a[j] = tanhf(s);
    }
    float res[3], pos[3];
    #pragma unroll
    for (int ii = 0; ii < 3; ++ii) {
        float s = 0.0f;
        #pragma unroll
        for (int j = 0; j < 3; ++j) s += a[j] * Wb[ii * 3 + j];
        res[ii] = s;
        pos[ii] = cons[n * 3 + ii] + s;
    }
    float pj0 = 0.0f, pj1 = 0.0f;
    #pragma unroll
    for (int ii = 0; ii < 3; ++ii) {
        pj0 += pos[ii] * r[b * 9 + ii * 3 + 0];
        pj1 += pos[ii] * r[b * 9 + ii * 3 + 1];
    }
    out_pp[(size_t)i * 2 + 0] = pj0;
    out_pp[(size_t)i * 2 + 1] = pj1;
    #pragma unroll
    for (int ii = 0; ii < 3; ++ii) {
        out_pos[(size_t)i * 3 + ii] = pos[ii];
        out_res[(size_t)i * 3 + ii] = res[ii];
    }
}

// ---------------------------------------------------------------------------
// Zero + bilinear splat with global f32 atomics
// ---------------------------------------------------------------------------
__global__ void k_zero(float* __restrict__ p, int n) {
    int i = blockIdx.x * blockDim.x + threadIdx.x;
    if (i < n) p[i] = 0.0f;
}

__global__ void k_splat(const float* __restrict__ pp, const float* __restrict__ amp,
                        const float* __restrict__ ampvar, float* __restrict__ im) {
    int i = blockIdx.x * blockDim.x + threadIdx.x;
    if (i >= MROWS) return;
    int b = i >> 14;
    int n = i & (NPTS - 1);
    // w = amp * softmax(ampvar, axis=0) over NCLS=2
    float a0 = ampvar[n], a1 = ampvar[NPTS + n];
    float mx = fmaxf(a0, a1);
    float e0 = __expf(a0 - mx), e1 = __expf(a1 - mx);
    float inv = amp[0] / (e0 + e1);
    float w0 = e0 * inv, w1 = e1 * inv;

    float px = pp[(size_t)i * 2 + 0], py = pp[(size_t)i * 2 + 1];
    float pex = fminf(fmaxf((px + 0.5f) * 255.0f, 0.0f), 254.9999f);
    float pey = fminf(fmaxf((py + 0.5f) * 255.0f, 0.0f), 254.9999f);
    int x0 = (int)floorf(pex), y0 = (int)floorf(pey);
    float fx = pex - (float)x0, fy = pey - (float)y0;
    int x1 = x0 + 1 < 255 ? x0 + 1 : 255;
    int y1 = y0 + 1 < 255 ? y0 + 1 : 255;
    float w00 = (1.0f - fx) * (1.0f - fy), w10 = fx * (1.0f - fy);
    float w01 = (1.0f - fx) * fy,          w11 = fx * fy;

    float* im0 = im + ((size_t)b * NCLS + 0) * SIM * SIM;
    float* im1 = im + ((size_t)b * NCLS + 1) * SIM * SIM;
    atomicAdd(im0 + y0 * SIM + x0, w0 * w00);
    atomicAdd(im0 + y0 * SIM + x1, w0 * w10);
    atomicAdd(im0 + y1 * SIM + x0, w0 * w01);
    atomicAdd(im0 + y1 * SIM + x1, w0 * w11);
    atomicAdd(im1 + y0 * SIM + x0, w1 * w00);
    atomicAdd(im1 + y0 * SIM + x1, w1 * w10);
    atomicAdd(im1 + y1 * SIM + x0, w1 * w01);
    atomicAdd(im1 + y1 * SIM + x1, w1 * w11);
}

// ---------------------------------------------------------------------------
// Radix-2 DIT 256-pt FFT in LDS. One block (128 threads) per line.
// ---------------------------------------------------------------------------
__device__ __forceinline__ void fft256_lds(float* re, float* im, int t) {
    for (int len = 2; len <= 256; len <<= 1) {
        int halfl = len >> 1;
        int g = t / halfl, p = t - g * halfl;
        int i = g * len + p, j = i + halfl;
        float ang = -2.0f * PI_F * (float)p / (float)len;
        float wr = __cosf(ang), wi = __sinf(ang);
        float xr = re[j], xi = im[j];
        float tr = wr * xr - wi * xi, ti = wr * xi + wi * xr;
        float ur = re[i], ui = im[i];
        re[i] = ur + tr; im[i] = ui + ti;
        re[j] = ur - tr; im[j] = ui - ti;
        __syncthreads();
    }
}

__global__ void k_fft_rows(const float* __restrict__ img_in, float2* __restrict__ X) {
    __shared__ float re[256], im[256];
    int line = blockIdx.x;                 // img*256 + row
    int t = threadIdx.x;                   // 0..127
    const float* src = img_in + (size_t)line * SIM;
    for (int idx = t; idx < 256; idx += 128) {
        int rsrc = __brev((unsigned)idx) >> 24;   // 8-bit bit reversal
        re[idx] = src[rsrc];
        im[idx] = 0.0f;
    }
    __syncthreads();
    fft256_lds(re, im, t);
    for (int idx = t; idx < 256; idx += 128)
        X[(size_t)line * SIM + idx] = make_float2(re[idx], im[idx]);
}

__global__ void k_fft_cols(float2* __restrict__ X) {
    __shared__ float re[256], im[256];
    int line = blockIdx.x;                 // img*256 + col
    int imgi = line >> 8, col = line & 255;
    int t = threadIdx.x;
    float2* base = X + (size_t)imgi * SIM * SIM + col;
    for (int idx = t; idx < 256; idx += 128) {
        int rsrc = __brev((unsigned)idx) >> 24;
        float2 v = base[(size_t)rsrc * SIM];
        re[idx] = v.x; im[idx] = v.y;
    }
    __syncthreads();
    fft256_lds(re, im, t);
    for (int idx = t; idx < 256; idx += 128)
        base[(size_t)idx * SIM] = make_float2(re[idx], im[idx]);
}

// ---------------------------------------------------------------------------
// Combine: Proj = shift( sum_c filt_c * Fim[:,c] )
// ---------------------------------------------------------------------------
__global__ void k_combine(const float2* __restrict__ X, const float* __restrict__ A_cls,
                          const float* __restrict__ B_cls, const float* __restrict__ shift,
                          float* __restrict__ proj_out) {
    int i = blockIdx.x * blockDim.x + threadIdx.x;
    if (i >= BATCH * SIM * SIM) return;
    int b = i >> 16;
    int y = (i >> 8) & 255;
    int x = i & 255;
    float ay = (y < 128) ? (float)y : (float)(y - 256);
    float ax = (x < 128) ? (float)x : (float)(x - 256);
    float R = sqrtf(ax * ax + ay * ay);
    float2 F0 = X[((size_t)b * 2 + 0) * SIM * SIM + (size_t)y * SIM + x];
    float2 F1 = X[((size_t)b * 2 + 1) * SIM * SIM + (size_t)y * SIM + x];
    float f0 = __expf(-B_cls[0] * B_cls[0] * R) * A_cls[0] * A_cls[0];
    float f1 = __expf(-B_cls[1] * B_cls[1] * R) * A_cls[1] * A_cls[1];
    float pr = f0 * F0.x + f1 * F1.x;
    float pi = f0 * F0.y + f1 * F1.y;
    float kx = ax * (1.0f / 256.0f), ky = ay * (1.0f / 256.0f);
    float ph = -2.0f * PI_F * (shift[b * 2 + 0] * kx + shift[b * 2 + 1] * ky);
    float cr = __cosf(ph), ci = __sinf(ph);
    proj_out[(size_t)i * 2 + 0] = pr * cr - pi * ci;
    proj_out[(size_t)i * 2 + 1] = pr * ci + pi * cr;
}

// ---------------------------------------------------------------------------
// Small MLP for v (8 rows) + radial mask gather for m
// ---------------------------------------------------------------------------
__global__ void k_vmlp(const float* __restrict__ z,
                       const float* __restrict__ Wv1, const float* __restrict__ bv1,
                       const float* __restrict__ Wv2, const float* __restrict__ bv2,
                       const float* __restrict__ Wv3, const float* __restrict__ bv3,
                       float* __restrict__ V3) {
    __shared__ float v1[128], v2[128];
    int t = threadIdx.x;
    for (int b = 0; b < BATCH; ++b) {
        float s = bv1[t];
        #pragma unroll
        for (int k = 0; k < 8; ++k) s += z[b * 8 + k] * Wv1[t * 8 + k];
        v1[t] = fmaxf(s, 0.0f);
        __syncthreads();
        float s2 = bv2[t];
        for (int k = 0; k < 128; ++k) s2 += v1[k] * Wv2[t * 128 + k];
        v2[t] = fmaxf(s2, 0.0f);
        __syncthreads();
        float s3 = bv3[t];
        for (int k = 0; k < 128; ++k) s3 += v2[k] * Wv3[t * 128 + k];
        V3[b * 128 + t] = 1.0f / (1.0f + __expf(-s3));
        __syncthreads();
    }
}

__global__ void k_mmask(const float* __restrict__ V3, float* __restrict__ m) {
    int i = blockIdx.x * blockDim.x + threadIdx.x;
    if (i >= BATCH * BOX * BOX) return;
    int b = i >> 14;
    int y = (i >> 7) & 127;
    int x = i & 127;
    float ay = (float)(y - 64), ax = (float)(x - 64);
    float R = sqrtf(ax * ax + ay * ay);
    int idx = (int)rintf(R);
    idx = idx < 0 ? 0 : (idx > 127 ? 127 : idx);
    m[i] = V3[b * 128 + idx];
}

// ---------------------------------------------------------------------------
// Launch
// ---------------------------------------------------------------------------
extern "C" void kernel_launch(void* const* d_in, const int* in_sizes, int n_in,
                              void* d_out, int out_size, void* d_ws, size_t ws_size,
                              hipStream_t stream) {
    (void)in_sizes; (void)n_in; (void)out_size; (void)ws_size;
    const float* z      = (const float*)d_in[0];
    const float* r      = (const float*)d_in[1];
    const float* cons   = (const float*)d_in[2];
    const float* amp    = (const float*)d_in[3];
    const float* ampvar = (const float*)d_in[4];
    const float* shift  = (const float*)d_in[5];
    const float* W0     = (const float*)d_in[6];
    const float* W1s    = (const float*)d_in[7];
    const float* W2s    = (const float*)d_in[8];
    const float* Wa     = (const float*)d_in[9];
    const float* Wb     = (const float*)d_in[10];
    const float* Wv1    = (const float*)d_in[11];
    const float* bv1    = (const float*)d_in[12];
    const float* Wv2    = (const float*)d_in[13];
    const float* bv2    = (const float*)d_in[14];
    const float* Wv3    = (const float*)d_in[15];
    const float* bv3    = (const float*)d_in[16];
    const float* A_cls  = (const float*)d_in[17];
    const float* B_cls  = (const float*)d_in[18];

    char* ws = (char*)d_ws;
    _Float16* W0p  = (_Float16*)(ws + WS_W0P);
    _Float16* W1h  = (_Float16*)(ws + WS_W1);
    _Float16* W2h  = (_Float16*)(ws + WS_W2);
    _Float16* FEAT = (_Float16*)(ws + WS_FEAT);
    _Float16* H    = (_Float16*)(ws + WS_H);
    _Float16* T    = (_Float16*)(ws + WS_T);
    float2*   X    = (float2*)(ws + WS_X);
    float*    V3   = (float*)(ws + WS_V3);

    float* out      = (float*)d_out;
    float* out_proj = out + O_PROJ;
    float* out_im   = out + O_IM;
    float* out_pp   = out + O_PP;
    float* out_pos  = out + O_POS;
    float* out_res  = out + O_RES;
    float* out_m    = out + O_M;

    // 1) Weights -> f16 (W0 padded K: 68 -> 96)
    k_f32_to_f16_pad<<<(128 * DIN_PAD + 255) / 256, 256, 0, stream>>>(W0, W0p, 128, 68, DIN_PAD);
    k_f32_to_f16<<<(NLAYERS * 128 * 128 + 255) / 256, 256, 0, stream>>>(W1s, W1h, NLAYERS * 128 * 128);
    k_f32_to_f16<<<(NLAYERS * 128 * 128 + 255) / 256, 256, 0, stream>>>(W2s, W2h, NLAYERS * 128 * 128);

    // 2) Feature builder (pos-enc + latent, zero-padded to K=96)
    k_feat<<<MROWS / 128, 128, 0, stream>>>(cons, z, FEAT);

    // 3) WMMA GEMM stack: h = feat @ W0^T ; 6x residual layers
    dim3 gblk(MROWS / MTILE_WAVE);   // 2048 blocks; each wave: 64x16 tile, 4 WMMAs per B-frag
    k_gemm_wmma<0><<<gblk, 256, 0, stream>>>(FEAT, W0p, H, nullptr, DIN_PAD);
    for (int l = 0; l < NLAYERS; ++l) {
        k_gemm_wmma<1><<<gblk, 256, 0, stream>>>(H, W1h + (size_t)l * 128 * 128, T, nullptr, 128);
        k_gemm_wmma<2><<<gblk, 256, 0, stream>>>(T, W2h + (size_t)l * 128 * 128, H, H, 128);
    }

    // 4) Head: tanh bottleneck, pos, rotated 2D projection
    k_head<<<MROWS / 128, 128, 0, stream>>>(H, cons, r, Wa, Wb, out_pp, out_pos, out_res);

    // 5) Splat (zero image then atomic bilinear scatter)
    k_zero<<<(BATCH * NCLS * SIM * SIM + 255) / 256, 256, 0, stream>>>(out_im, BATCH * NCLS * SIM * SIM);
    k_splat<<<MROWS / 128, 128, 0, stream>>>(out_pp, amp, ampvar, out_im);

    // 6) 2D FFT (rows then cols), per (b, class) image
    k_fft_rows<<<BATCH * NCLS * SIM, 128, 0, stream>>>(out_im, X);
    k_fft_cols<<<BATCH * NCLS * SIM, 128, 0, stream>>>(X);

    // 7) CTF filter + class sum + Fourier shift -> Proj (complex interleaved)
    k_combine<<<(BATCH * SIM * SIM + 255) / 256, 256, 0, stream>>>(X, A_cls, B_cls, shift, out_proj);

    // 8) v-MLP and radial-index mask gather -> m
    k_vmlp<<<1, 128, 0, stream>>>(z, Wv1, bv1, Wv2, bv2, Wv3, bv3, V3);
    k_mmask<<<(BATCH * BOX * BOX + 255) / 256, 256, 0, stream>>>(V3, out_m);
}
